// Model_79972291051642
// MI455X (gfx1250) — matrix operand order repaired and checked by reading, not compiled, not run on hardware
//
#include <hip/hip_runtime.h>
#include <math.h>

// ---------------------------------------------------------------------------
// CDNA5 (gfx1250) wave32 WMMA helpers.  Layouts (A/B contiguous k, C/D per CDNA5 ISA 7.12.2):
//  A 16x32 f16 : lanes 0-15 -> rows 0-15, halves 0..15 = K[kc+0..15];
//                lanes 16-31 -> same rows, K[kc+16..31].
//  B 32x16 f16 : lanes 0-15 -> col N, halves 0..15 = K[kc+0..15];
//                lanes 16-31 -> same cols, K[kc+16..31].
//  C/D 16x16 f32: lane L -> col (L&15); VGPR v -> row v (+8 for lanes>=16).
// B matrices are pre-transposed/padded to f16 in ws: Bt[col*KPAD + k],
// so a B fragment is 16 contiguous halves (32B aligned) -> 2x b128 loads.
// ---------------------------------------------------------------------------
typedef __attribute__((ext_vector_type(16))) _Float16 v16h;
typedef __attribute__((ext_vector_type(8)))  _Float16 v8h;
typedef __attribute__((ext_vector_type(8)))  float    v8f;

__device__ __forceinline__ v16h frag_split(const _Float16* p, int gap) {
    v8h lo = *(const v8h*)p;
    v8h hi = *(const v8h*)(p + gap);
    return __builtin_shufflevector(lo, hi, 0, 1, 2, 3, 4, 5, 6, 7,
                                   8, 9, 10, 11, 12, 13, 14, 15);
}

// A fragment from LDS tile with row stride ASTR (halves).
__device__ __forceinline__ v16h a_frag(const _Float16* As, int astr, int kc, int lane) {
    const int r   = lane & 15;
    const int hi8 = (lane >> 4) << 4;          // 0 or 16
    return frag_split(As + r * astr + kc + hi8, 8);
}

// B fragment from pre-transposed f16 weights (stride = KPAD halves).
__device__ __forceinline__ v16h b_frag(const _Float16* Bt, int kpad, int kc, int c0, int lane) {
    const int col = c0 + (lane & 15);
    const int kb  = kc + ((lane >> 4) << 4);   // 0 or 16
    return *(const v16h*)(Bt + col * kpad + kb);
}

__device__ __forceinline__ v8f wmma_f16(v16h a, v16h b, v8f c) {
    return __builtin_amdgcn_wmma_f32_16x16x32_f16(false, a, false, b, (short)0, c,
                                                  false, false);
}

// ---------------------------------------------------------------------------
// One-time weight prep: Bt[col*Kpad + k] = f16(W[k*Ncols + col]), zero padded.
// ---------------------------------------------------------------------------
__global__ void prep_weight_kernel(const float* __restrict__ W, _Float16* __restrict__ Bt,
                                   int Kact, int Kpad, int Ncols, int Npad) {
    int i = blockIdx.x * blockDim.x + threadIdx.x;
    if (i >= Npad * Kpad) return;
    int col = i / Kpad, kk = i - col * Kpad;
    float v = (col < Ncols && kk < Kact) ? W[kk * Ncols + col] : 0.0f;
    Bt[i] = (_Float16)v;
}

// ---------------------------------------------------------------------------
// Tiled GEMM:  C[M x N] = act(A[M x KPAD] * Bt^T + bias)
// Block: NW waves, MT row-tiles of 16.  A staged to LDS as f16 with padded
// stride; B fragments read straight from pre-transposed f16 global weights.
// ---------------------------------------------------------------------------
template <int KPAD, int NW, int MT>
__global__ void gemm_kernel(const float* __restrict__ A, const _Float16* __restrict__ Bt,
                            float* __restrict__ C, int Mrows, int Ncols,
                            const float* __restrict__ bias, int do_relu) {
    constexpr int ASTR = KPAD + 8;             // bank-conflict-free row stride
    __shared__ _Float16 As[MT * 16 * ASTR];
    const int tid  = threadIdx.x;
    const int lane = tid & 31;
    const int wave = tid >> 5;
    const int r0   = blockIdx.x * (16 * MT);

    for (int i = tid; i < MT * 16 * KPAD; i += NW * 32) {
        int r  = i / KPAD;
        int kk = i - r * KPAD;
        float v = (r0 + r < Mrows) ? A[(size_t)(r0 + r) * KPAD + kk] : 0.0f;
        As[r * ASTR + kk] = (_Float16)v;
    }
    __syncthreads();

    v8f acc[MT];
#pragma unroll
    for (int mt = 0; mt < MT; ++mt) acc[mt] = (v8f){};

    const int c0 = wave * 16;
#pragma unroll
    for (int kc = 0; kc < KPAD; kc += 32) {
        v16h b = b_frag(Bt, KPAD, kc, c0, lane);
#pragma unroll
        for (int mt = 0; mt < MT; ++mt) {
            v16h a = a_frag(As + mt * 16 * ASTR, ASTR, kc, lane);
            acc[mt] = wmma_f16(a, b, acc[mt]);
        }
    }

    const int col = c0 + (lane & 15);
    const int rb  = (lane >> 4) << 3;
    if (col < Ncols) {
        const float bv = bias ? bias[col] : 0.0f;
#pragma unroll
        for (int mt = 0; mt < MT; ++mt) {
#pragma unroll
            for (int v = 0; v < 8; ++v) {
                int r = r0 + mt * 16 + rb + v;
                if (r < Mrows) {
                    float y = acc[mt][v] + bv;
                    if (do_relu) y = fmaxf(y, 0.0f);
                    C[(size_t)r * Ncols + col] = y;
                }
            }
        }
    }
}

// ---------------------------------------------------------------------------
// Brute-force KNN, one query per thread, refs tiled through LDS.
// Fully-unrolled insertion ladder keeps the top-k list in VGPRs.
// ---------------------------------------------------------------------------
#define KNN_TILE 256

template <int KK>
__global__ void knn_kernel(const float* __restrict__ qp, const float* __restrict__ rp,
                           int nq, int nr, int* __restrict__ out) {
    __shared__ float tile[KNN_TILE * 3];
    const int tid = threadIdx.x;
    const int qi  = blockIdx.x * blockDim.x + tid;

    float qx = 0.f, qy = 0.f, qz = 0.f;
    if (qi < nq) { qx = qp[qi * 3 + 0]; qy = qp[qi * 3 + 1]; qz = qp[qi * 3 + 2]; }

    float best[KK];
    int   bid[KK];
#pragma unroll
    for (int j = 0; j < KK; ++j) { best[j] = 3.4e38f; bid[j] = 0; }

    for (int t0 = 0; t0 < nr; t0 += KNN_TILE) {
        const int cnt = min(KNN_TILE, nr - t0);
        __syncthreads();
        for (int i = tid; i < cnt * 3; i += blockDim.x) tile[i] = rp[(size_t)t0 * 3 + i];
        // speculative prefetch of the next ref tile -> global_prefetch_b8
        if (t0 + KNN_TILE < nr && tid < min(KNN_TILE, nr - t0 - KNN_TILE))
            __builtin_prefetch(rp + (size_t)(t0 + KNN_TILE) * 3 + tid * 3, 0, 0);
        __syncthreads();
        for (int c = 0; c < cnt; ++c) {
            float dx = tile[c * 3 + 0] - qx;
            float dy = tile[c * 3 + 1] - qy;
            float dz = tile[c * 3 + 2] - qz;
            float d  = fmaf(dx, dx, fmaf(dy, dy, dz * dz));
            if (d < best[KK - 1]) {
                best[KK - 1] = d;
                bid[KK - 1]  = t0 + c;
#pragma unroll
                for (int j = KK - 1; j > 0; --j) {
                    if (best[j] < best[j - 1]) {
                        float tf = best[j]; best[j] = best[j - 1]; best[j - 1] = tf;
                        int   ti = bid[j];  bid[j]  = bid[j - 1];  bid[j - 1]  = ti;
                    }
                }
            }
        }
    }
    if (qi < nq) {
#pragma unroll
        for (int j = 0; j < KK; ++j) out[(size_t)qi * KK + j] = bid[j];
    }
}

// ---------------------------------------------------------------------------
// Fused gather + MLP(9->64->64) + max-pool over K=16 neighbors.
// One point per block; 4 waves; Wg2 matmul via WMMA (A = 16 neighbors x 64).
// ---------------------------------------------------------------------------
__global__ void gather_mlp_kernel(const float* __restrict__ p0, const float* __restrict__ x0,
                                  const int* __restrict__ idx, const float* __restrict__ Wg1,
                                  const _Float16* __restrict__ Wg2t, float* __restrict__ x1) {
    __shared__ int      nb[16];
    __shared__ float    g[16][9];
    __shared__ _Float16 h1[16 * 72];
    const int m    = blockIdx.x;
    const int tid  = threadIdx.x;
    const int lane = tid & 31;
    const int wave = tid >> 5;

    if (tid < 16) nb[tid] = idx[m * 16 + tid];
    __syncthreads();
    if (tid < 144) {
        int k = tid / 9, j = tid - k * 9;
        int n = nb[k];
        g[k][j] = (j < 3) ? (p0[n * 3 + j] - p0[m * 3 + j]) : x0[n * 6 + (j - 3)];
    }
    __syncthreads();
    for (int i = tid; i < 1024; i += blockDim.x) {
        int k = i >> 6, c = i & 63;
        float s = 0.f;
#pragma unroll
        for (int j = 0; j < 9; ++j) s = fmaf(g[k][j], Wg1[j * 64 + c], s);
        h1[k * 72 + c] = (_Float16)fmaxf(s, 0.0f);
    }
    __syncthreads();

    v8f acc = {};
#pragma unroll
    for (int kc = 0; kc < 64; kc += 32) {
        v16h a = a_frag(h1, 72, kc, lane);
        v16h b = b_frag(Wg2t, 64, kc, wave * 16, lane);
        acc = wmma_f16(a, b, acc);
    }
    // relu + max over neighbor rows (8 rows in this lane, 8 in partner lane)
    float pm = 0.0f;
#pragma unroll
    for (int v = 0; v < 8; ++v) pm = fmaxf(pm, fmaxf(acc[v], 0.0f));
    pm = fmaxf(pm, __shfl_xor(pm, 16, 32));
    if (lane < 16) x1[(size_t)m * 64 + wave * 16 + lane] = pm;
}

// ---------------------------------------------------------------------------
// Fused vector-attention block per point m (K=16 neighbors).
// pe = relu(rel@Wpe1)@Wpe2 (WMMA); a = relu((q-k[idx]+pe)@Wa1) (WMMA) @Wa2;
// softmax over K; grouped weighted sum of (v+pe).
// ---------------------------------------------------------------------------
__global__ void attention_kernel(const float* __restrict__ p1, const int* __restrict__ idx2,
                                 const float* __restrict__ qf, const float* __restrict__ kf,
                                 const float* __restrict__ vf, const float* __restrict__ Wpe1,
                                 const _Float16* __restrict__ Wpe2t,
                                 const _Float16* __restrict__ Wa1t,
                                 const float* __restrict__ Wa2, float* __restrict__ xb) {
    __shared__ int      nb[16];
    __shared__ float    rel[16][3];
    __shared__ float    qm[64];
    __shared__ _Float16 pein[16 * 72];
    __shared__ float    pef[16 * 64];
    __shared__ _Float16 ain[16 * 72];
    __shared__ _Float16 a1s[16 * 64];
    __shared__ float    a2s[16 * 8];
    const int m    = blockIdx.x;
    const int tid  = threadIdx.x;
    const int lane = tid & 31;
    const int wave = tid >> 5;

    if (tid < 16) nb[tid] = idx2[m * 16 + tid];
    if (tid >= 64 && tid < 128) qm[tid - 64] = qf[(size_t)m * 64 + (tid - 64)];
    __syncthreads();
    if (tid < 48) {
        int k = tid / 3, j = tid - k * 3;
        rel[k][j] = p1[nb[k] * 3 + j] - p1[m * 3 + j];
    }
    __syncthreads();
    for (int i = tid; i < 1024; i += blockDim.x) {
        int k = i >> 6, c = i & 63;
        float s = fmaf(rel[k][0], Wpe1[c],
                  fmaf(rel[k][1], Wpe1[64 + c], rel[k][2] * Wpe1[128 + c]));
        pein[k * 72 + c] = (_Float16)fmaxf(s, 0.0f);
    }
    __syncthreads();

    {   // pe = pein @ Wpe2
        v8f acc = {};
#pragma unroll
        for (int kc = 0; kc < 64; kc += 32) {
            v16h a = a_frag(pein, 72, kc, lane);
            v16h b = b_frag(Wpe2t, 64, kc, wave * 16, lane);
            acc = wmma_f16(a, b, acc);
        }
        int col = wave * 16 + (lane & 15);
        int rb  = (lane >> 4) << 3;
#pragma unroll
        for (int v = 0; v < 8; ++v) pef[(rb + v) * 64 + col] = acc[v];
    }
    __syncthreads();
    for (int i = tid; i < 1024; i += blockDim.x) {
        int k = i >> 6, c = i & 63;
        ain[k * 72 + c] = (_Float16)(qm[c] - kf[(size_t)nb[k] * 64 + c] + pef[i]);
    }
    __syncthreads();

    {   // a1 = relu(ain @ Wa1)
        v8f acc = {};
#pragma unroll
        for (int kc = 0; kc < 64; kc += 32) {
            v16h a = a_frag(ain, 72, kc, lane);
            v16h b = b_frag(Wa1t, 64, kc, wave * 16, lane);
            acc = wmma_f16(a, b, acc);
        }
        int col = wave * 16 + (lane & 15);
        int rb  = (lane >> 4) << 3;
#pragma unroll
        for (int v = 0; v < 8; ++v)
            a1s[(rb + v) * 64 + col] = (_Float16)fmaxf(acc[v], 0.0f);
    }
    __syncthreads();
    if (tid < 128) {   // a2 = a1 @ Wa2  (64 -> 8)
        int k = tid >> 3, s = tid & 7;
        float acc = 0.f;
        for (int c = 0; c < 64; ++c) acc = fmaf((float)a1s[k * 64 + c], Wa2[c * 8 + s], acc);
        a2s[k * 8 + s] = acc;
    }
    __syncthreads();
    if (tid < 8) {     // softmax over K (axis=1) per share-column
        int s = tid;
        float mx = -3.4e38f;
        for (int k = 0; k < 16; ++k) mx = fmaxf(mx, a2s[k * 8 + s]);
        float sum = 0.f;
        for (int k = 0; k < 16; ++k) { float e = expf(a2s[k * 8 + s] - mx); a2s[k * 8 + s] = e; sum += e; }
        float inv = 1.0f / sum;
        for (int k = 0; k < 16; ++k) a2s[k * 8 + s] *= inv;
    }
    __syncthreads();
    if (tid < 64) {    // grouped aggregation: weight index = c % 8
        int c = tid;
        float acc = 0.f;
        for (int k = 0; k < 16; ++k)
            acc = fmaf(vf[(size_t)nb[k] * 64 + c] + pef[k * 64 + c], a2s[k * 8 + (c & 7)], acc);
        xb[(size_t)m * 64 + c] = acc;
    }
}

// ---------------------------------------------------------------------------
// Batch-norm statistics (mean / biased var) per channel; one channel per block.
// ---------------------------------------------------------------------------
__global__ void stats_kernel(const float* __restrict__ X, int R, int C,
                             float* __restrict__ mean, float* __restrict__ var) {
    __shared__ float ss[256], sq[256];
    const int ch  = blockIdx.x;
    const int tid = threadIdx.x;
    float s = 0.f, q = 0.f;
    for (int i = tid; i < R; i += 256) {
        float v = X[(size_t)i * C + ch];
        s += v; q = fmaf(v, v, q);
    }
    ss[tid] = s; sq[tid] = q;
    __syncthreads();
    for (int off = 128; off > 0; off >>= 1) {
        if (tid < off) { ss[tid] += ss[tid + off]; sq[tid] += sq[tid + off]; }
        __syncthreads();
    }
    if (tid == 0) {
        float mu = ss[0] / (float)R;
        mean[ch] = mu;
        var[ch]  = sq[0] / (float)R - mu * mu;
    }
}

__global__ void bn_apply_kernel(const float* __restrict__ X, const float* __restrict__ mean,
                                const float* __restrict__ var, const float* __restrict__ g,
                                const float* __restrict__ b, const float* __restrict__ resid,
                                int do_relu, float* __restrict__ Y, int R, int C) {
    int i = blockIdx.x * blockDim.x + threadIdx.x;
    if (i >= R * C) return;
    int c = i % C;
    float y = (X[i] - mean[c]) * rsqrtf(var[c] + 1e-5f) * g[c] + b[c];
    if (resid)   y += resid[i];
    if (do_relu) y = fmaxf(y, 0.0f);
    Y[i] = y;
}

// ---------------------------------------------------------------------------
// 3-NN inverse-distance interpolation; writes padded feature block
// F0[n, 0:64]=xi, [64:70]=x0[n], [70:96]=0.  One point per block (96 threads).
// ---------------------------------------------------------------------------
__global__ void interp_kernel(const float* __restrict__ p0, const int* __restrict__ idx3,
                              const float* __restrict__ xc, const float* __restrict__ x0,
                              float* __restrict__ F0) {
    const int n = blockIdx.x;
    const int c = threadIdx.x;
    int i0 = idx3[n * 3 + 0], i1 = idx3[n * 3 + 1], i2 = idx3[n * 3 + 2];
    float px = p0[n * 3 + 0], py = p0[n * 3 + 1], pz = p0[n * 3 + 2];
    float w0, w1, w2;
    {
        float dx, dy, dz, d;
        dx = px - p0[i0 * 3]; dy = py - p0[i0 * 3 + 1]; dz = pz - p0[i0 * 3 + 2];
        d = fmaf(dx, dx, fmaf(dy, dy, dz * dz)); w0 = 1.0f / (d + 1e-8f);
        dx = px - p0[i1 * 3]; dy = py - p0[i1 * 3 + 1]; dz = pz - p0[i1 * 3 + 2];
        d = fmaf(dx, dx, fmaf(dy, dy, dz * dz)); w1 = 1.0f / (d + 1e-8f);
        dx = px - p0[i2 * 3]; dy = py - p0[i2 * 3 + 1]; dz = pz - p0[i2 * 3 + 2];
        d = fmaf(dx, dx, fmaf(dy, dy, dz * dz)); w2 = 1.0f / (d + 1e-8f);
        float inv = 1.0f / (w0 + w1 + w2);
        w0 *= inv; w1 *= inv; w2 *= inv;
    }
    if (c < 64) {
        float v = w0 * xc[(size_t)i0 * 64 + c] + w1 * xc[(size_t)i1 * 64 + c] +
                  w2 * xc[(size_t)i2 * 64 + c];
        F0[(size_t)n * 96 + c] = v;
    } else {
        F0[(size_t)n * 96 + c] = (c < 70) ? x0[(size_t)n * 6 + (c - 64)] : 0.0f;
    }
}

// ---------------------------------------------------------------------------
// Host-side orchestration
// ---------------------------------------------------------------------------
extern "C" void kernel_launch(void* const* d_in, const int* in_sizes, int n_in,
                              void* d_out, int out_size, void* d_ws, size_t ws_size,
                              hipStream_t stream) {
    (void)in_sizes; (void)n_in; (void)out_size; (void)ws_size;
    const int N = 32768, M = 8192;

    const float* p0   = (const float*)d_in[0];
    const float* x0   = (const float*)d_in[1];
    const float* Wg1  = (const float*)d_in[2];
    const float* Wg2  = (const float*)d_in[3];
    const float* Wl1  = (const float*)d_in[4];
    const float* g1   = (const float*)d_in[5];
    const float* b1   = (const float*)d_in[6];
    const float* Wq   = (const float*)d_in[7];
    const float* Wk   = (const float*)d_in[8];
    const float* Wv   = (const float*)d_in[9];
    const float* Wpe1 = (const float*)d_in[10];
    const float* Wpe2 = (const float*)d_in[11];
    const float* Wa1  = (const float*)d_in[12];
    const float* Wa2  = (const float*)d_in[13];
    const float* g2   = (const float*)d_in[14];
    const float* b2   = (const float*)d_in[15];
    const float* Wl3  = (const float*)d_in[16];
    const float* g3   = (const float*)d_in[17];
    const float* b3   = (const float*)d_in[18];
    const float* Wfp1 = (const float*)d_in[19];
    const float* Wfp2 = (const float*)d_in[20];
    const float* Wc1  = (const float*)d_in[21];
    const float* bc1  = (const float*)d_in[22];
    const float* gc   = (const float*)d_in[23];
    const float* bc   = (const float*)d_in[24];
    const float* Wc2  = (const float*)d_in[25];
    const float* bc2  = (const float*)d_in[26];
    float* out = (float*)d_out;

    char*  w   = (char*)d_ws;
    size_t off = 0;
    auto alloc = [&](size_t bytes) -> void* {
        void* p = w + off;
        off += (bytes + 255) & ~(size_t)255;
        return p;
    };
    int*   idx1 = (int*)  alloc((size_t)M * 16 * 4);
    int*   idx2 = (int*)  alloc((size_t)M * 16 * 4);
    int*   idx3 = (int*)  alloc((size_t)N * 3 * 4);
    float* x1   = (float*)alloc((size_t)M * 64 * 4);
    float* tA   = (float*)alloc((size_t)M * 64 * 4);
    float* xa   = (float*)alloc((size_t)M * 64 * 4);
    float* qb   = (float*)alloc((size_t)M * 64 * 4);
    float* kb   = (float*)alloc((size_t)M * 64 * 4);
    float* vb   = (float*)alloc((size_t)M * 64 * 4);
    float* xb   = (float*)alloc((size_t)M * 64 * 4);
    float* xc   = (float*)alloc((size_t)M * 64 * 4);
    float* F0   = (float*)alloc((size_t)N * 96 * 4);
    float* f1   = (float*)alloc((size_t)N * 128 * 4);
    float* f2   = (float*)alloc((size_t)N * 128 * 4);
    float* t2   = (float*)alloc((size_t)N * 128 * 4);
    float* mean = (float*)alloc(128 * 4);
    float* var  = (float*)alloc(128 * 4);
    // pre-transposed f16 weights
    _Float16* Wg2t  = (_Float16*)alloc(64 * 64 * 2);
    _Float16* Wl1t  = (_Float16*)alloc(64 * 64 * 2);
    _Float16* Wqt   = (_Float16*)alloc(64 * 64 * 2);
    _Float16* Wkt   = (_Float16*)alloc(64 * 64 * 2);
    _Float16* Wvt   = (_Float16*)alloc(64 * 64 * 2);
    _Float16* Wpe2t = (_Float16*)alloc(64 * 64 * 2);
    _Float16* Wa1t  = (_Float16*)alloc(64 * 64 * 2);
    _Float16* Wl3t  = (_Float16*)alloc(64 * 64 * 2);
    _Float16* Wfp1t = (_Float16*)alloc(128 * 96 * 2);
    _Float16* Wfp2t = (_Float16*)alloc(128 * 128 * 2);
    _Float16* Wc1t  = (_Float16*)alloc(128 * 128 * 2);
    _Float16* Wc2t  = (_Float16*)alloc(16 * 128 * 2);

    // ---- weight prep (one-time, L2-resident) ----
    auto prep = [&](const float* W, _Float16* Bt, int Kact, int Kpad, int Nc, int Np) {
        int tot = Np * Kpad;
        prep_weight_kernel<<<(tot + 255) / 256, 256, 0, stream>>>(W, Bt, Kact, Kpad, Nc, Np);
    };
    prep(Wg2, Wg2t, 64, 64, 64, 64);
    prep(Wl1, Wl1t, 64, 64, 64, 64);
    prep(Wq,  Wqt,  64, 64, 64, 64);
    prep(Wk,  Wkt,  64, 64, 64, 64);
    prep(Wv,  Wvt,  64, 64, 64, 64);
    prep(Wpe2, Wpe2t, 64, 64, 64, 64);
    prep(Wa1,  Wa1t,  64, 64, 64, 64);
    prep(Wl3,  Wl3t,  64, 64, 64, 64);
    prep(Wfp1, Wfp1t, 70, 96, 128, 128);
    prep(Wfp2, Wfp2t, 128, 128, 128, 128);
    prep(Wc1,  Wc1t,  128, 128, 128, 128);
    prep(Wc2,  Wc2t,  128, 128, 13, 16);

    // 1) idx = knn(p1, p0, 16)   (p1 == first M rows of p0)
    knn_kernel<16><<<(M + 255) / 256, 256, 0, stream>>>(p0, p0, M, N, idx1);
    // 2) x1 = maxpool_k relu(relu([rel|x0[idx]] @ Wg1) @ Wg2)
    gather_mlp_kernel<<<M, 128, 0, stream>>>(p0, x0, idx1, Wg1, Wg2t, x1);
    // 3) xa = relu(bn(x1 @ Wl1, g1, b1))
    gemm_kernel<64, 4, 4><<<M / 64, 128, 0, stream>>>(x1, Wl1t, tA, M, 64, nullptr, 0);
    stats_kernel<<<64, 256, 0, stream>>>(tA, M, 64, mean, var);
    bn_apply_kernel<<<(M * 64 + 255) / 256, 256, 0, stream>>>(tA, mean, var, g1, b1,
                                                              nullptr, 1, xa, M, 64);
    // 4) idx2 = knn(p1, p1, 16)
    knn_kernel<16><<<(M + 255) / 256, 256, 0, stream>>>(p0, p0, M, M, idx2);
    // 5) q, k, v projections
    gemm_kernel<64, 4, 4><<<M / 64, 128, 0, stream>>>(xa, Wqt, qb, M, 64, nullptr, 0);
    gemm_kernel<64, 4, 4><<<M / 64, 128, 0, stream>>>(xa, Wkt, kb, M, 64, nullptr, 0);
    gemm_kernel<64, 4, 4><<<M / 64, 128, 0, stream>>>(xa, Wvt, vb, M, 64, nullptr, 0);
    // 6) fused vector attention -> xb
    attention_kernel<<<M, 128, 0, stream>>>(p0, idx2, qb, kb, vb, Wpe1, Wpe2t, Wa1t, Wa2, xb);
    // 7) xb = relu(bn(xb, g2, b2))   (in place)
    stats_kernel<<<64, 256, 0, stream>>>(xb, M, 64, mean, var);
    bn_apply_kernel<<<(M * 64 + 255) / 256, 256, 0, stream>>>(xb, mean, var, g2, b2,
                                                              nullptr, 1, xb, M, 64);
    // 8) xc = relu(bn(xb @ Wl3, g3, b3) + x1)
    gemm_kernel<64, 4, 4><<<M / 64, 128, 0, stream>>>(xb, Wl3t, tA, M, 64, nullptr, 0);
    stats_kernel<<<64, 256, 0, stream>>>(tA, M, 64, mean, var);
    bn_apply_kernel<<<(M * 64 + 255) / 256, 256, 0, stream>>>(tA, mean, var, g3, b3,
                                                              x1, 1, xc, M, 64);
    // 9) idx3 = knn(p0, p1, 3); F0 = [interp(xc) | x0 | pad]
    knn_kernel<3><<<(N + 255) / 256, 256, 0, stream>>>(p0, p0, N, M, idx3);
    interp_kernel<<<N, 96, 0, stream>>>(p0, idx3, xc, x0, F0);
    // 10) f1 = relu(F0 @ Wfp1)  (K=70 padded to 96)
    gemm_kernel<96, 8, 4><<<N / 64, 256, 0, stream>>>(F0, Wfp1t, f1, N, 128, nullptr, 1);
    // 11) f2 = relu(f1 @ Wfp2)
    gemm_kernel<128, 8, 4><<<N / 64, 256, 0, stream>>>(f1, Wfp2t, f2, N, 128, nullptr, 1);
    // 12) hc = relu(bn(f2 @ Wc1 + bc1, gc, bc))  (hc reuses f1)
    gemm_kernel<128, 8, 4><<<N / 64, 256, 0, stream>>>(f2, Wc1t, t2, N, 128, bc1, 0);
    stats_kernel<<<128, 256, 0, stream>>>(t2, N, 128, mean, var);
    bn_apply_kernel<<<(N * 128 + 255) / 256, 256, 0, stream>>>(t2, mean, var, gc, bc,
                                                               nullptr, 1, f1, N, 128);
    // 13) out = hc @ Wc2 + bc2  (N=13 cols, one wave per tile)
    gemm_kernel<128, 1, 4><<<N / 64, 32, 0, stream>>>(f1, Wc2t, out, N, 13, bc2, 0);
}
